// Attention_54254026883611
// MI455X (gfx1250) — compile-verified, matrix operand order checked by reading, not run on hardware
//
#include <hip/hip_runtime.h>
#include <hip/hip_bf16.h>

#define NXC   1024
#define NQKV  3072
#define TC    2048
#define BC    2
#define HC    16
#define HDC   64
#define MROWS (BC*TC)   // 4096

typedef __bf16 bf16;
typedef __bf16 v16bf  __attribute__((ext_vector_type(16)));
typedef __bf16 bf16x8 __attribute__((ext_vector_type(8)));
typedef __bf16 bf16x4 __attribute__((ext_vector_type(4)));
typedef float  v8f    __attribute__((ext_vector_type(8)));
typedef unsigned int u32x4 __attribute__((ext_vector_type(4)));
typedef int          i32x4 __attribute__((ext_vector_type(4)));
typedef int          i32x8 __attribute__((ext_vector_type(8)));

#if defined(__has_builtin)
#  if __has_builtin(__builtin_amdgcn_tensor_load_to_lds) && \
      __has_builtin(__builtin_amdgcn_s_wait_tensorcnt)
#    define USE_TDM 1
#  else
#    define USE_TDM 0
#  endif
#else
#  define USE_TDM 0
#endif

__device__ __forceinline__ v8f wmma_bf16(v16bf a, v16bf b, v8f c) {
  // (neg_a, A, neg_b, B, c_mod, C, reuse_a, reuse_b)
  return __builtin_amdgcn_wmma_f32_16x16x32_bf16(false, a, false, b, (short)0, c,
                                                 false, false);
}

// Load a 16x32 bf16 operand fragment from a row-major tile (stride in elems,
// multiple of 8 for 16B alignment). A operand: `row` = M index. B operand:
// pass the NxK-transposed tile with `row` = N index. Per ISA layout lanes<16
// hold K {0..7, 16..23}, lanes>=16 hold K {8..15, 24..31} -> two 16B loads.
__device__ __forceinline__ v16bf load_frag(const bf16* base, int row, int stride,
                                           int lane) {
  const int kb = (lane & 16) ? 8 : 0;
  const bf16* p = base + row * stride + kb;
  bf16x8 lo = *(const bf16x8*)(p);
  bf16x8 hi = *(const bf16x8*)(p + 16);
  return __builtin_shufflevector(lo, hi, 0, 1, 2, 3, 4, 5, 6, 7,
                                 8, 9, 10, 11, 12, 13, 14, 15);
}

// ---------------------------------------------------------------------------
// Kernel 1: qkv = x @ w_attn + b_attn, fused split-heads scatter to bf16
// Q/K/V (B, H, T, HD). Block tile 128(M) x 128(N), 8 waves (4x2); each wave
// 32x64 = 2x4 WMMA fragments (8 WMMAs per K-step of 32).
// ---------------------------------------------------------------------------
__global__ __launch_bounds__(256)
void qkv_gemm(const float* __restrict__ x, const float* __restrict__ w,
              const float* __restrict__ bias,
              bf16* __restrict__ Q, bf16* __restrict__ K, bf16* __restrict__ V) {
  __shared__ bf16 As[128 * 40];   // 128 M rows x 32 K (pad->40: 80B rows)
  __shared__ bf16 Bs[128 * 40];   // W^T tile: 128 N rows x 32 K

  const int tid  = threadIdx.x;
  const int lane = tid & 31;
  const int wave = tid >> 5;
  const int m0 = blockIdx.x * 128;
  const int n0 = blockIdx.y * 128;
  const int wm = (wave >> 1) * 32;
  const int wn = (wave & 1) * 64;

  const v8f vz = {0.f, 0.f, 0.f, 0.f, 0.f, 0.f, 0.f, 0.f};
  v8f acc[8] = {vz, vz, vz, vz, vz, vz, vz, vz};

  for (int kbi = 0; kbi < NXC; kbi += 32) {
    // Stage A (x) tile fp32->bf16: 8 threads/row, 4 floats each.
#pragma unroll
    for (int pass = 0; pass < 4; ++pass) {
      int row = pass * 32 + (tid >> 3);
      int c4  = (tid & 7) * 4;
      float4 xv = *(const float4*)(x + (size_t)(m0 + row) * NXC + kbi + c4);
      bf16x4 bv = {(bf16)xv.x, (bf16)xv.y, (bf16)xv.z, (bf16)xv.w};
      *(bf16x4*)(&As[row * 40 + c4]) = bv;
    }
    // Stage W tile transposed (N x K), coalesced reads along N (128 wide).
#pragma unroll
    for (int pass = 0; pass < 4; ++pass) {
      int k  = pass * 8 + (tid >> 5);
      int nn = (tid & 31) * 4;
      float4 wv = *(const float4*)(w + (size_t)(kbi + k) * NQKV + n0 + nn);
      Bs[(nn + 0) * 40 + k] = (bf16)wv.x;
      Bs[(nn + 1) * 40 + k] = (bf16)wv.y;
      Bs[(nn + 2) * 40 + k] = (bf16)wv.z;
      Bs[(nn + 3) * 40 + k] = (bf16)wv.w;
    }
    // Prefetch next K-step tiles into cache (global_prefetch_b8).
    if (kbi + 32 < NXC) {
      __builtin_prefetch(x + (size_t)(m0 + (tid >> 3)) * NXC + kbi + 32, 0, 0);
      __builtin_prefetch(w + (size_t)(kbi + 32 + (tid >> 5)) * NQKV + n0 +
                             (tid & 31) * 4, 0, 0);
    }
    __syncthreads();

    v16bf a0 = load_frag(As, wm + (lane & 15), 40, lane);
    v16bf a1 = load_frag(As, wm + 16 + (lane & 15), 40, lane);
#pragma unroll
    for (int ni = 0; ni < 4; ++ni) {
      v16bf bfr = load_frag(Bs, wn + ni * 16 + (lane & 15), 40, lane);
      acc[ni]     = wmma_bf16(a0, bfr, acc[ni]);
      acc[4 + ni] = wmma_bf16(a1, bfr, acc[4 + ni]);
    }
    __syncthreads();
  }

  // Epilogue: bias + split-heads scatter. C/D layout: row 8*(lane>=16)+r,
  // col lane&15.
#pragma unroll
  for (int f = 0; f < 8; ++f) {
    const int mi = f >> 2, ni = f & 3;
    const int n = n0 + wn + ni * 16 + (lane & 15);
    const float bv = bias[n];
    const int which = n >> 10;           // 0=q, 1=k, 2=v
    const int nx = n & (NXC - 1);
    const int hh = nx >> 6;
    const int d  = nx & 63;
    bf16* dst = (which == 0) ? Q : (which == 1) ? K : V;
#pragma unroll
    for (int r = 0; r < 8; ++r) {
      int m = m0 + wm + mi * 16 + ((lane >> 4) << 3) + r;
      int bb = m >> 11;                  // / T
      int t  = m & (TC - 1);
      float val = acc[f][r] + bv;
      dst[((size_t)((bb * HC + hh) * TC + t)) * HDC + d] = (bf16)val;
    }
  }
}

// ---------------------------------------------------------------------------
// Kernel 2: causal flash attention per (b, h, 64-row q-tile). 4 waves, each
// wave owns 16 q rows. K tile DMAed by the Tensor Data Mover (with LDS pad
// producing the 72-elem row stride); V staged transposed; online softmax in
// fp32 registers; P round-trips LDS to convert C-layout -> A-layout.
// ---------------------------------------------------------------------------
__global__ __launch_bounds__(128)
void flash_attn(const bf16* __restrict__ Q, const bf16* __restrict__ K,
                const bf16* __restrict__ V, bf16* __restrict__ Aout) {
  __shared__ bf16 Ks[64 * 72];      // K tile rows (kk x d), 72 = 64 + 8 pad
  __shared__ bf16 Vt[64 * 72];      // V transposed (d x kk)
  __shared__ bf16 Ps[4][16 * 72];   // per-wave P staging (q x kk)

  const int tid  = threadIdx.x;
  const int lane = tid & 31;
  const int wave = tid >> 5;
  const int qt = blockIdx.x, h = blockIdx.y, b = blockIdx.z;

  const bf16* Qbh = Q + (size_t)((b * HC + h) * TC) * HDC;
  const bf16* Kbh = K + (size_t)((b * HC + h) * TC) * HDC;
  const bf16* Vbh = V + (size_t)((b * HC + h) * TC) * HDC;
  const int qw = qt * 64 + wave * 16;

  // Q fragments for this wave's 16 rows (d = 0..31, 32..63), kept in regs.
  v16bf qf0 = load_frag(Qbh + (size_t)qw * HDC, (lane & 15), HDC, lane);
  v16bf qf1 = load_frag(Qbh + (size_t)qw * HDC + 32, (lane & 15), HDC, lane);

  const v8f vz = {0.f, 0.f, 0.f, 0.f, 0.f, 0.f, 0.f, 0.f};
  v8f o[4] = {vz, vz, vz, vz};
  float mrow[8], lrow[8];
#pragma unroll
  for (int r = 0; r < 8; ++r) { mrow[r] = -1e30f; lrow[r] = 0.f; }

  const int nkt = qt + 1;               // causal: tiles 0..qt
  for (int kt = 0; kt < nkt; ++kt) {
    const int kbase = kt * 64;
    __syncthreads();                    // protect Ks/Vt/Ps from prior iter

#if USE_TDM
    // TDM: DMA the 64x64 bf16 K tile into LDS with padding (4 DWORDs after
    // every 32 DWORDs -> 72-element rows). D# per CDNA5 ISA.
    if (wave == 0) {
      unsigned long long ga =
          (unsigned long long)(const void*)(Kbh + (size_t)kbase * HDC);
      u32x4 g0 = {1u,                                  // count=1 descriptor
                  (unsigned)(size_t)(&Ks[0]),          // lds_addr
                  (unsigned)ga,                        // global_addr[31:0]
                  ((unsigned)(ga >> 32) & 0x01FFFFFFu) | (2u << 30)}; // +type=2
      i32x8 g1 = {(int)0x07110000u, // data_size=2B, pad_en, interval=32dw, amt=4dw
                  (int)(64u << 16), // tensor_dim0 = 64
                  (int)(64u << 16), // tensor_dim1 = 64
                  (int)(64u << 16), // tile_dim0 = 64
                  64,               // tile_dim1 = 64
                  64,               // tensor_dim0_stride = 64
                  0, 0};
      i32x4 z4 = {0, 0, 0, 0};
#if __clang_major__ >= 23
      i32x8 z8 = {0, 0, 0, 0, 0, 0, 0, 0};
      __builtin_amdgcn_tensor_load_to_lds(g0, g1, z4, z4, z8, 0);
#else
      __builtin_amdgcn_tensor_load_to_lds(g0, g1, z4, z4, 0);
#endif
    }
#endif

#pragma unroll
    for (int pass = 0; pass < 4; ++pass) {
      int row = pass * 16 + (tid >> 3);
      int seg = (tid & 7) * 8;
#if !USE_TDM
      bf16x8 kv = *(const bf16x8*)(Kbh + (size_t)(kbase + row) * HDC + seg);
      *(bf16x8*)(&Ks[row * 72 + seg]) = kv;
#endif
      bf16x8 vv = *(const bf16x8*)(Vbh + (size_t)(kbase + row) * HDC + seg);
#pragma unroll
      for (int i = 0; i < 8; ++i) Vt[(seg + i) * 72 + row] = vv[i];
    }
    // Prefetch next K/V tiles.
    if (kt + 1 < nkt) {
      __builtin_prefetch(Kbh + (size_t)(kbase + 64 + (tid >> 1)) * HDC, 0, 0);
      __builtin_prefetch(Vbh + (size_t)(kbase + 64 + (tid >> 1)) * HDC, 0, 0);
    }
#if USE_TDM
    if (wave == 0) __builtin_amdgcn_s_wait_tensorcnt(0);
#endif
    __syncthreads();

    // S = Q K^T for this wave's 16 q rows x 64 kk cols (4 fragments).
    v8f s[4];
#pragma unroll
    for (int j = 0; j < 4; ++j) {
      v8f sj = vz;
      v16bf k0 = load_frag(Ks, j * 16 + (lane & 15), 72, lane);
      v16bf k1 = load_frag(Ks + 32, j * 16 + (lane & 15), 72, lane);
      sj = wmma_bf16(qf0, k0, sj);
      sj = wmma_bf16(qf1, k1, sj);
      s[j] = sj;
    }

    // scale 1/sqrt(64) + causal mask
    const int qr_base = qw + ((lane >> 4) << 3);
    const int kc_base = kbase + (lane & 15);
#pragma unroll
    for (int j = 0; j < 4; ++j) {
      const int kc = kc_base + j * 16;
#pragma unroll
      for (int r = 0; r < 8; ++r) {
        float sv = s[j][r] * 0.125f;
        s[j][r] = (kc > qr_base + r) ? -1e30f : sv;
      }
    }

    // online softmax (rows live on 16-lane half-groups)
#pragma unroll
    for (int r = 0; r < 8; ++r) {
      float mx = fmaxf(fmaxf(s[0][r], s[1][r]), fmaxf(s[2][r], s[3][r]));
#pragma unroll
      for (int off = 1; off < 16; off <<= 1)
        mx = fmaxf(mx, __shfl_xor(mx, off, 32));
      const float mold = mrow[r];
      const float mnew = fmaxf(mold, mx);
      const float corr = __expf(mold - mnew);
      mrow[r] = mnew;
      float rsum = 0.f;
#pragma unroll
      for (int j = 0; j < 4; ++j) {
        float p = __expf(s[j][r] - mnew);
        rsum += p;
        Ps[wave][(((lane >> 4) << 3) + r) * 72 + j * 16 + (lane & 15)] = (bf16)p;
      }
#pragma unroll
      for (int off = 1; off < 16; off <<= 1)
        rsum += __shfl_xor(rsum, off, 32);
      lrow[r] = lrow[r] * corr + rsum;
#pragma unroll
      for (int ds = 0; ds < 4; ++ds) o[ds][r] *= corr;
    }
    __syncthreads();                    // make Ps visible for A-layout reload

    // O += P V  (A = P in A-layout from LDS, B = V^T rows, d x kk)
    v16bf p0 = load_frag(Ps[wave], (lane & 15), 72, lane);
    v16bf p1 = load_frag(Ps[wave] + 32, (lane & 15), 72, lane);
#pragma unroll
    for (int ds = 0; ds < 4; ++ds) {
      v16bf v0 = load_frag(Vt, ds * 16 + (lane & 15), 72, lane);
      v16bf v1 = load_frag(Vt + 32, ds * 16 + (lane & 15), 72, lane);
      o[ds] = wmma_bf16(p0, v0, o[ds]);
      o[ds] = wmma_bf16(p1, v1, o[ds]);
    }
  }

  // Normalize and write merged-head (B, T, NX) bf16 for the proj GEMM.
#pragma unroll
  for (int r = 0; r < 8; ++r) {
    const float linv = 1.0f / lrow[r];
    const int qrow = qw + ((lane >> 4) << 3) + r;
#pragma unroll
    for (int ds = 0; ds < 4; ++ds) {
      const int d = ds * 16 + (lane & 15);
      Aout[(size_t)(b * TC + qrow) * NXC + h * HDC + d] = (bf16)(o[ds][r] * linv);
    }
  }
}

// ---------------------------------------------------------------------------
// Kernel 3: out = A @ w_proj + b_proj  (A is bf16 from kernel 2, out fp32).
// Same 128x128 tiling as kernel 1.
// ---------------------------------------------------------------------------
__global__ __launch_bounds__(256)
void proj_gemm(const bf16* __restrict__ A, const float* __restrict__ w,
               const float* __restrict__ bias, float* __restrict__ out) {
  __shared__ bf16 As[128 * 40];
  __shared__ bf16 Bs[128 * 40];

  const int tid  = threadIdx.x;
  const int lane = tid & 31;
  const int wave = tid >> 5;
  const int m0 = blockIdx.x * 128;
  const int n0 = blockIdx.y * 128;
  const int wm = (wave >> 1) * 32;
  const int wn = (wave & 1) * 64;

  const v8f vz = {0.f, 0.f, 0.f, 0.f, 0.f, 0.f, 0.f, 0.f};
  v8f acc[8] = {vz, vz, vz, vz, vz, vz, vz, vz};

  for (int kbi = 0; kbi < NXC; kbi += 32) {
    // A already bf16: 4 threads/row, 8 elems (16B) each.
#pragma unroll
    for (int pass = 0; pass < 2; ++pass) {
      int row = pass * 64 + (tid >> 2);
      int c8  = (tid & 3) * 8;
      bf16x8 av = *(const bf16x8*)(A + (size_t)(m0 + row) * NXC + kbi + c8);
      *(bf16x8*)(&As[row * 40 + c8]) = av;
    }
#pragma unroll
    for (int pass = 0; pass < 4; ++pass) {
      int k  = pass * 8 + (tid >> 5);
      int nn = (tid & 31) * 4;
      float4 wv = *(const float4*)(w + (size_t)(kbi + k) * NXC + n0 + nn);
      Bs[(nn + 0) * 40 + k] = (bf16)wv.x;
      Bs[(nn + 1) * 40 + k] = (bf16)wv.y;
      Bs[(nn + 2) * 40 + k] = (bf16)wv.z;
      Bs[(nn + 3) * 40 + k] = (bf16)wv.w;
    }
    if (kbi + 32 < NXC) {
      __builtin_prefetch(A + (size_t)(m0 + (tid >> 2)) * NXC + kbi + 32, 0, 0);
      __builtin_prefetch(w + (size_t)(kbi + 32 + (tid >> 5)) * NXC + n0 +
                             (tid & 31) * 4, 0, 0);
    }
    __syncthreads();

    v16bf a0 = load_frag(As, wm + (lane & 15), 40, lane);
    v16bf a1 = load_frag(As, wm + 16 + (lane & 15), 40, lane);
#pragma unroll
    for (int ni = 0; ni < 4; ++ni) {
      v16bf bfr = load_frag(Bs, wn + ni * 16 + (lane & 15), 40, lane);
      acc[ni]     = wmma_bf16(a0, bfr, acc[ni]);
      acc[4 + ni] = wmma_bf16(a1, bfr, acc[4 + ni]);
    }
    __syncthreads();
  }

#pragma unroll
  for (int f = 0; f < 8; ++f) {
    const int mi = f >> 2, ni = f & 3;
    const int n = n0 + wn + ni * 16 + (lane & 15);
    const float bv = bias[n];
#pragma unroll
    for (int r = 0; r < 8; ++r) {
      int m = m0 + wm + mi * 16 + ((lane >> 4) << 3) + r;
      out[(size_t)m * NXC + n] = acc[f][r] + bv;
    }
  }
}

// ---------------------------------------------------------------------------
extern "C" void kernel_launch(void* const* d_in, const int* in_sizes, int n_in,
                              void* d_out, int out_size, void* d_ws, size_t ws_size,
                              hipStream_t stream) {
  (void)in_sizes; (void)n_in; (void)out_size; (void)ws_size;
  const float* x      = (const float*)d_in[0];
  const float* w_attn = (const float*)d_in[1];
  const float* b_attn = (const float*)d_in[2];
  const float* w_proj = (const float*)d_in[3];
  const float* b_proj = (const float*)d_in[4];
  float* out = (float*)d_out;

  const size_t nper = (size_t)BC * HC * TC * HDC;   // 4M elems
  bf16* Q = (bf16*)d_ws;
  bf16* K = Q + nper;
  bf16* V = K + nper;
  bf16* A = V + nper;                                // total 32 MB of ws

  dim3 g1(MROWS / 128, NQKV / 128);
  qkv_gemm<<<g1, 256, 0, stream>>>(x, w_attn, b_attn, Q, K, V);

  dim3 g2(TC / 64, HC, BC);
  flash_attn<<<g2, 128, 0, stream>>>(Q, K, V, A);

  dim3 g3(MROWS / 128, NXC / 128);
  proj_gemm<<<g3, 256, 0, stream>>>(A, w_proj, b_proj, out);
}
